// DynamicDRFConv2d_463856468653
// MI455X (gfx1250) — matrix operand order repaired and checked by reading, not compiled, not run on hardware
//
#include <hip/hip_runtime.h>

typedef __attribute__((ext_vector_type(16))) _Float16 v16h;
typedef __attribute__((ext_vector_type(8)))  _Float16 v8h;
typedef __attribute__((ext_vector_type(8)))  float    v8f;

#define B_    16
#define C_    256
#define W_    56
#define HW    3136      // 56*56
#define HID   65
#define KSEL  4
#define OUTC  256
#define KKDIM 2304      // C_*9
#define NWEL  589824    // OUTC*KKDIM

#define ASTRIDE 40      // LDS row stride in halves (80B: 16B-aligned chunks, low conflict)
#define BSTRIDE 40

// workspace layout (byte offsets)
#define WS_POOLED  0        // 4096 floats
#define WS_MAXSLOT 16384    // 1 uint (float bits)
#define WS_SEL     16448    // 16 ints
#define WS_WQ      32768    // 4*NWEL _Float16  (~4.7 MB)

#if __has_builtin(__builtin_amdgcn_global_load_async_to_lds_b128) && \
    __has_builtin(__builtin_amdgcn_s_wait_asynccnt)
#define USE_ASYNC_LDS 1
// per hipcc diagnostic: param0 is AS(1) pointer to 'int __attribute__((vector_size(16)))'
typedef int v4i_t __attribute__((__vector_size__(4 * sizeof(int))));
typedef __attribute__((address_space(1))) v4i_t* gptr_v4i;
typedef __attribute__((address_space(3))) v4i_t* lptr_v4i;
#else
#define USE_ASYNC_LDS 0
#endif

// ---------------- K1: adaptive avg-pool (mean over HW) ----------------
__global__ __launch_bounds__(256) void pool_kernel(const float* __restrict__ x,
                                                   float* __restrict__ pooled,
                                                   unsigned* __restrict__ maxslot) {
  __shared__ float red[256];
  const int bc = blockIdx.x;          // b*C + c
  const int tid = threadIdx.x;
  const float* p = x + (size_t)bc * HW;
  float s = 0.f;
  for (int i = tid; i < HW; i += 256) s += p[i];
  red[tid] = s;
  __syncthreads();
  for (int off = 128; off > 0; off >>= 1) {
    if (tid < off) red[tid] += red[tid + off];
    __syncthreads();
  }
  if (tid == 0) pooled[bc] = red[0] * (1.f / 3136.f);
  if (bc == 0 && tid == 0) *maxslot = 0u;   // init for K3 (stream-ordered)
}

// ---------------- K2: MLP attention -> raw + argmax selection ----------------
__global__ __launch_bounds__(256) void mlp_kernel(const float* __restrict__ pooled,
                                                  const float* __restrict__ w_fc1,
                                                  const float* __restrict__ w_fc2,
                                                  const float* __restrict__ b_fc2,
                                                  float* __restrict__ raw_out,
                                                  int* __restrict__ sel) {
  __shared__ float hbuf[B_ * HID];
  __shared__ float rawbuf[B_ * KSEL];
  const int tid = threadIdx.x;
  for (int idx = tid; idx < B_ * HID; idx += 256) {
    const int b = idx / HID, j = idx - b * HID;
    float acc = 0.f;
    for (int c = 0; c < C_; ++c) acc += pooled[b * C_ + c] * w_fc1[j * C_ + c];
    hbuf[idx] = fmaxf(acc, 0.f);
  }
  __syncthreads();
  if (tid < B_ * KSEL) {
    const int b = tid >> 2, k = tid & 3;
    float acc = b_fc2[k];
    for (int j = 0; j < HID; ++j) acc += hbuf[b * HID + j] * w_fc2[k * HID + j];
    rawbuf[tid] = acc;
    raw_out[tid] = acc;           // raw tuple output, row-major [B,K]
  }
  __syncthreads();
  if (tid < B_) {
    float best = rawbuf[tid * 4];
    int bi = 0;
    for (int k = 1; k < 4; ++k) {
      const float v = rawbuf[tid * 4 + k];
      if (v > best) { best = v; bi = k; }   // first-max tie-break like argmax
    }
    sel[tid] = bi;
  }
}

// ---------------- K3: global max |tanh(w)| ----------------
__global__ __launch_bounds__(256) void maxabs_kernel(const float* __restrict__ w,
                                                     unsigned* __restrict__ slot, int n) {
  __shared__ float red[256];
  const int tid = threadIdx.x;
  float m = 0.f;
  for (int i = blockIdx.x * 256 + tid; i < n; i += gridDim.x * 256)
    m = fmaxf(m, fabsf(tanhf(w[i])));
  red[tid] = m;
  __syncthreads();
  for (int off = 128; off > 0; off >>= 1) {
    if (tid < off) red[tid] = fmaxf(red[tid], red[tid + off]);
    __syncthreads();
  }
  if (tid == 0) atomicMax(slot, __float_as_uint(red[0]));  // vals >= 0 -> monotone bits
}

// ---------------- K4: DoReFa quantize -> 4 f16 banks, tap-major K layout ----------------
// dst[bank][o][tap][c]  <-  src weight[o][c][ky][kx], tap = ky*3+kx
__global__ __launch_bounds__(256) void quant_kernel(const float* __restrict__ w,
                                                    const unsigned* __restrict__ slot,
                                                    _Float16* __restrict__ wq, int n) {
  const float inv = 1.f / __uint_as_float(*slot);
  for (int j = blockIdx.x * 256 + threadIdx.x; j < n; j += gridDim.x * 256) {
    const int o   = j / KKDIM;
    const int rr  = j - o * KKDIM;
    const int tap = rr >> 8;       // rr / 256
    const int c   = rr & 255;
    const float t  = tanhf(w[o * KKDIM + c * 9 + tap]);
    const float xq = t * inv * 0.5f + 0.5f;   // in [0,1]
    #pragma unroll
    for (int bank = 0; bank < 4; ++bank) {
      const float a = (float)(4 << bank);     // bits 2..5
      const float q = fminf(floorf(a * xq), a - 1.f) * (1.f / a);
      wq[(size_t)bank * NWEL + j] = (_Float16)(q * 2.f - 1.f);  // exact in f16
    }
  }
}

// ---------------- K5: per-sample implicit-GEMM conv via WMMA ----------------
// Block = 256 thr (8 waves). Tile: M=256 (all out-ch) x N=64 (spatial).
// K = tap-major (9 taps x 256 ch): tap outer (predicate hoisted), 8 x 32-ch K-steps inner.
// Wave (4x2) computes 64x32 = 4x2 wmma_f32_16x16x32_f16 tiles -> 8 WMMA / K-step.
__global__ __launch_bounds__(256) void conv_wmma_kernel(const float* __restrict__ x,
                                                        const _Float16* __restrict__ wq,
                                                        const float* __restrict__ bias,
                                                        const int* __restrict__ sel,
                                                        float* __restrict__ out) {
  __shared__ _Float16 Asm[256 * ASTRIDE];   // [o][k_local] (20.5 KB)
  __shared__ _Float16 Bsm[64 * BSTRIDE];    // [s_local][k_local] (5 KB)

  const int b   = blockIdx.y;
  const int n0  = blockIdx.x * 64;
  const int tid = threadIdx.x;
  const int ks  = sel[b];

  const _Float16* Ag = wq + (size_t)ks * NWEL;     // [o][tap*256+c]
  const float*    xb = x + (size_t)b * C_ * HW;

  // B-gather coords: thread fills 8 consecutive channels for one spatial point
  const int s_local = tid & 63;
  const int cb0     = (tid >> 6) * 8;
  const int s       = n0 + s_local;
  const int sy      = s / W_;
  const int sx      = s - sy * W_;

  // wave / lane decomposition (wave32)
  const int lane = tid & 31;
  const int lm   = lane & 15;
  const int hi   = lane >> 4;
  const int wave = tid >> 5;
  const int wm   = (wave >> 1) * 64;   // 0,64,128,192
  const int wn   = (wave & 1) * 32;    // 0,32

  v8f acc[4][2] = {};

  for (int tap = 0; tap < 9; ++tap) {
    const int ky = tap / 3;
    const int kx = tap - 3 * ky;
    const int iy = sy + ky - 1;
    const int ix = sx + kx - 1;
    const bool valid = ((unsigned)iy < 56u) && ((unsigned)ix < 56u);
    const float* xt = xb + (valid ? (iy * W_ + ix) : 0);   // clamped: always safe to read

    for (int c0 = 0; c0 < C_; c0 += 32) {
      // ---- stage A tile: thread tid copies its o-row's 32 halves (4 x 16B) ----
      {
        const uint4* gs = reinterpret_cast<const uint4*>(Ag + (size_t)tid * KKDIM + tap * C_ + c0);
        uint4* ds = reinterpret_cast<uint4*>(&Asm[tid * ASTRIDE]);
#if USE_ASYNC_LDS
        #pragma unroll
        for (int i = 0; i < 4; ++i)
          __builtin_amdgcn_global_load_async_to_lds_b128(
              (gptr_v4i)(gs + i), (lptr_v4i)(ds + i), 0, 0);
#else
        #pragma unroll
        for (int i = 0; i < 4; ++i) ds[i] = gs[i];
#endif
      }
      // ---- stage B tile: gather 8 channels, pack, single b128 store ----
      {
        v8h pack;
        #pragma unroll
        for (int i = 0; i < 8; ++i) {
          const int c = c0 + cb0 + i;
          pack[i] = (_Float16)(valid ? xt[c * HW] : 0.f);
        }
        *reinterpret_cast<v8h*>(&Bsm[s_local * BSTRIDE + cb0]) = pack;
      }
#if USE_ASYNC_LDS
      __builtin_amdgcn_s_wait_asynccnt(0);
#endif
      __syncthreads();

      // ---- fragments (ISA 16-bit layouts), all 16B-aligned b128 chunks ----
      v16h afrag[4];
      #pragma unroll
      for (int sm = 0; sm < 4; ++sm) {
        const _Float16* ap = &Asm[(wm + sm * 16 + lm) * ASTRIDE + hi * 8];
        const v8h lo = *reinterpret_cast<const v8h*>(ap);        // K = hi*8 + 0..7
        const v8h hv = *reinterpret_cast<const v8h*>(ap + 16);   // K = 16 + hi*8 + 0..7
        afrag[sm] = __builtin_shufflevector(lo, hv, 0,1,2,3,4,5,6,7,8,9,10,11,12,13,14,15);
      }
      v16h bfrag[2];
      #pragma unroll
      for (int sn = 0; sn < 2; ++sn) {
        const _Float16* bp = &Bsm[(wn + sn * 16 + lm) * BSTRIDE + hi * 8];
        const v8h lo = *reinterpret_cast<const v8h*>(bp);        // K = hi*8 + 0..7
        const v8h hv = *reinterpret_cast<const v8h*>(bp + 16);   // K = 16 + hi*8 + 0..7
        bfrag[sn] = __builtin_shufflevector(lo, hv, 0,1,2,3,4,5,6,7,8,9,10,11,12,13,14,15);
      }

      #pragma unroll
      for (int sm = 0; sm < 4; ++sm)
        #pragma unroll
        for (int sn = 0; sn < 2; ++sn)
          acc[sm][sn] = __builtin_amdgcn_wmma_f32_16x16x32_f16(
              false, afrag[sm], false, bfrag[sn], (short)0, acc[sm][sn], false, false);
      __syncthreads();
    }
  }

  // ---- epilogue: C/D layout lane col n=lm, VGPR r -> M = r + 8*hi; add bias ----
  const float* bb = bias + ks * OUTC;
  #pragma unroll
  for (int sm = 0; sm < 4; ++sm) {
    #pragma unroll
    for (int sn = 0; sn < 2; ++sn) {
      const int scol = n0 + wn + sn * 16 + lm;
      #pragma unroll
      for (int r = 0; r < 8; ++r) {
        const int o = wm + sm * 16 + r + 8 * hi;
        out[((size_t)b * OUTC + o) * HW + scol] = acc[sm][sn][r] + bb[o];
      }
    }
  }
}

extern "C" void kernel_launch(void* const* d_in, const int* in_sizes, int n_in,
                              void* d_out, int out_size, void* d_ws, size_t ws_size,
                              hipStream_t stream) {
  const float* x      = (const float*)d_in[0];
  const float* w_fc1  = (const float*)d_in[1];
  const float* w_fc2  = (const float*)d_in[2];
  const float* b_fc2  = (const float*)d_in[3];
  const float* weight = (const float*)d_in[4];
  const float* bias   = (const float*)d_in[5];

  float* out = (float*)d_out;
  char*  ws  = (char*)d_ws;
  float*     pooled  = (float*)(ws + WS_POOLED);
  unsigned*  maxslot = (unsigned*)(ws + WS_MAXSLOT);
  int*       sel     = (int*)(ws + WS_SEL);
  _Float16*  wq      = (_Float16*)(ws + WS_WQ);
  float*     raw_out = out + (size_t)B_ * OUTC * HW;   // tuple tail: raw [16,4]

  pool_kernel  <<<B_ * C_, 256, 0, stream>>>(x, pooled, maxslot);
  mlp_kernel   <<<1,       256, 0, stream>>>(pooled, w_fc1, w_fc2, b_fc2, raw_out, sel);
  maxabs_kernel<<<576,     256, 0, stream>>>(weight, maxslot, NWEL);
  quant_kernel <<<1152,    256, 0, stream>>>(weight, maxslot, wq, NWEL);
  conv_wmma_kernel<<<dim3(49, B_), 256, 0, stream>>>(x, wq, bias, sel, out);
}
